// MultiheadAttention_12532714569824
// MI455X (gfx1250) — compile-verified, matrix operand order checked by reading, not run on hardware
//
#include <hip/hip_runtime.h>

// MHA forward for B=4, T=2048, D=1024, H=16, Hd=64 on gfx1250.
// bf16 WMMA everywhere; fp32->bf16 hoisted to a bandwidth-bound pre-pass.
// Block-cooperative flash attention: one block = 8 waves on 8 consecutive
// 32-query tiles of the SAME (b,h); each 32-key K/V tile is async-staged
// into LDS once per block (double-buffered) and shared by all waves.
// Staging uses GLOBAL_LOAD_ASYNC_TO_LDS_B128 + s_wait_asynccnt (builtin or
// inline-asm fallback).

#define B_   4
#define T_   2048
#define D_   1024
#define H_   16
#define HD_  64
#define SCALE_ 0.125f   // 1/sqrt(64)

typedef __attribute__((ext_vector_type(16))) __bf16 bfrag;   // A/B operand: 8 VGPRs
typedef __attribute__((ext_vector_type(8)))  __bf16 bhalf8;  // 16-byte chunk
typedef __attribute__((ext_vector_type(8)))  float  v8f;     // C/D accumulator
typedef __attribute__((ext_vector_type(4)))  int    v4i;     // async-copy unit

#define WMMA_BF16(A_, B_m, C_) \
  __builtin_amdgcn_wmma_f32_16x16x32_bf16(false, (A_), false, (B_m), (short)0, (C_), false, false)

// A-fragment per-lane element mapping (16-bit, 16x32): lane half = lane>>4,
// elements 0..7  -> K = half*8 .. half*8+7
// elements 8..15 -> K = 16+half*8 .. 16+half*8+7
// => two contiguous 8-element (16B) chunks. B mirrors this with lane = column.

__device__ __forceinline__ bfrag frag_from_bf16(const __bf16* p0, const __bf16* p1) {
  bhalf8 a = *(const bhalf8*)p0;
  bhalf8 b = *(const bhalf8*)p1;
  return __builtin_shufflevector(a, b, 0,1,2,3,4,5,6,7,8,9,10,11,12,13,14,15);
}

__device__ __forceinline__ float redmax16(float v) {
  v = fmaxf(v, __shfl_xor(v, 1, 32));
  v = fmaxf(v, __shfl_xor(v, 2, 32));
  v = fmaxf(v, __shfl_xor(v, 4, 32));
  v = fmaxf(v, __shfl_xor(v, 8, 32));
  return v;
}
__device__ __forceinline__ float redsum16(float v) {
  v += __shfl_xor(v, 1, 32);
  v += __shfl_xor(v, 2, 32);
  v += __shfl_xor(v, 4, 32);
  v += __shfl_xor(v, 8, 32);
  return v;
}

// ---------------------------------------------------------------------------
// Async global -> LDS staging (16 bytes per call per thread).
// Builtin signature (from hipcc diagnostics): (AS1 v4i*, AS3 v4i*, imm, cpol).
// ---------------------------------------------------------------------------
#if __has_builtin(__builtin_amdgcn_global_load_async_to_lds_b128)
#define HAVE_ASYNC_LDS 1
#else
#define HAVE_ASYNC_LDS 0
#endif

typedef __attribute__((address_space(1))) v4i* gv4i_p;
typedef __attribute__((address_space(3))) v4i* lv4i_p;

__device__ __forceinline__ void stage16(const __bf16* g, __bf16* l) {
#if HAVE_ASYNC_LDS
  __builtin_amdgcn_global_load_async_to_lds_b128(
      (gv4i_p)(size_t)g, (lv4i_p)(size_t)l, 0, 0);
#else
  *(bhalf8*)l = *(const bhalf8*)g;     // register-relay fallback
#endif
}

__device__ __forceinline__ void stage_sync() {
#if HAVE_ASYNC_LDS
#if __has_builtin(__builtin_amdgcn_s_wait_asynccnt)
  __builtin_amdgcn_s_wait_asynccnt(0);
#else
  asm volatile("s_wait_asynccnt 0" ::: "memory");   // SOPP opcode 74
#endif
#endif
  __syncthreads();
}

// ---------------------------------------------------------------------------
// Bulk f32 -> bf16 conversion (bandwidth-bound pre-pass).
// ---------------------------------------------------------------------------
__global__ void f32_to_bf16_kernel(const float* __restrict__ in,
                                   __bf16* __restrict__ out, int n8) {
  int i = blockIdx.x * blockDim.x + threadIdx.x;
  if (i >= n8) return;
  const float4* p = (const float4*)in + (size_t)i * 2;
  float4 a = p[0], b = p[1];
  bhalf8 o;
  o[0] = (__bf16)a.x; o[1] = (__bf16)a.y; o[2] = (__bf16)a.z; o[3] = (__bf16)a.w;
  o[4] = (__bf16)b.x; o[5] = (__bf16)b.y; o[6] = (__bf16)b.z; o[7] = (__bf16)b.w;
  *((bhalf8*)out + i) = o;
}

// ---------------------------------------------------------------------------
// QKV projection: out = X @ W^T  (X: [B*T, D] bf16, W: [D, D] bf16 row-major)
// Wave computes a 16(M) x 128(N) tile.
//   vtrans==0 : out[((b*H+h)*T + t)*HD + hd]      (Q, K)
//   vtrans==1 : out[((b*H+h)*HD + hd)*T + t]      (V transposed for PV GEMM)
// ---------------------------------------------------------------------------
__global__ void qkv_proj_kernel(const __bf16* __restrict__ X,
                                const __bf16* __restrict__ W,
                                __bf16* __restrict__ out, int vtrans) {
  const int lane  = threadIdx.x & 31;
  const int lanec = lane & 15;
  const int half  = lane >> 4;
  const int rowb  = half * 8;

  const int wid   = blockIdx.x * 8 + (threadIdx.x >> 5);   // 4096 wave tasks
  const int n128  = wid & 7;
  const int mtile = wid >> 3;
  const int row0  = mtile * 16;

  v8f acc[8] = {};
  const __bf16* xrow  = X + (size_t)(row0 + lanec) * D_;
  const __bf16* wbase = W + (size_t)(n128 * 128 + lanec) * D_;

  for (int kb = 0; kb < D_ / 32; ++kb) {
    const int kbase = kb * 32;
    bfrag a = frag_from_bf16(xrow + kbase + half * 8,
                             xrow + kbase + 16 + half * 8);
    bfrag bf[8];
#pragma unroll
    for (int nt = 0; nt < 8; ++nt) {
      const __bf16* wrow = wbase + (size_t)nt * 16 * D_ + kbase;
      bf[nt] = frag_from_bf16(wrow + half * 8, wrow + 16 + half * 8);
    }
#pragma unroll
    for (int nt = 0; nt < 8; ++nt)
      acc[nt] = WMMA_BF16(a, bf[nt], acc[nt]);
  }

  const int bb = row0 >> 11;
  const int t0 = row0 & (T_ - 1);
#pragma unroll
  for (int nt = 0; nt < 8; ++nt) {
    const int nf = n128 * 128 + nt * 16 + lanec;
    const int h  = nf >> 6;
    const int hd = nf & 63;
#pragma unroll
    for (int g = 0; g < 8; ++g) {
      const int t = t0 + rowb + g;
      size_t idx = vtrans
        ? ((size_t)((bb * H_ + h) * HD_ + hd) * T_ + t)
        : ((size_t)((bb * H_ + h) * T_ + t) * HD_ + hd);
      out[idx] = (__bf16)acc[nt][g];
    }
  }
}

// ---------------------------------------------------------------------------
// Block-cooperative flash attention.
// grid: 64 (b,h) x 8 query-groups = 512 blocks; block = 8 waves = 256 thr.
// Q,K: [B*H, T, HD] bf16.  Vt: [B*H, HD, T] bf16.  Out: [B*T, D] bf16.
// ---------------------------------------------------------------------------
__global__ void flash_attn_kernel(const __bf16* __restrict__ Q,
                                  const __bf16* __restrict__ K,
                                  const __bf16* __restrict__ Vt,
                                  __bf16* __restrict__ Oout) {
  __shared__ __bf16 kbuf[2][32 * HD_];        // 2 x 4 KiB : 32 keys x 64 hd
  __shared__ __bf16 vbuf[2][HD_ * 32];        // 2 x 4 KiB : 64 hd x 32 keys
  __shared__ __bf16 psm[8][16 * 32];          // per-wave P tile (C->A relayout)

  const int tid   = threadIdx.x;
  const int wslot = tid >> 5;
  __bf16* sm = psm[wslot];

  const int lane  = tid & 31;
  const int lanec = lane & 15;
  const int half  = lane >> 4;
  const int rowb  = half * 8;

  const int bh     = blockIdx.x >> 3;          // 64 heads total
  const int qgrp   = blockIdx.x & 7;           // 8 query groups of 256
  const int qtw    = qgrp * 8 + wslot;         // this wave's 32-query tile
  const int q0     = qtw * 32;
  const int ktmax  = qgrp * 8 + 7;             // block-uniform loop bound
  const size_t bhT = (size_t)bh * T_;

  // Per-thread staging source addresses (advance by 32 keys per tile).
  const int skey = tid >> 3, sch8 = tid & 7;   // K tile: 32 x (8 chunks)
  const int shd  = tid >> 2, sch4 = tid & 3;   // V tile: 64 x (4 chunks)
  const __bf16* ksrc = K + (bhT + skey) * HD_ + sch8 * 8;
  const __bf16* vsrc = Vt + ((size_t)bh * HD_ + shd) * T_ + sch4 * 8;
  __bf16* kdst[2] = { kbuf[0] + skey * HD_ + sch8 * 8, kbuf[1] + skey * HD_ + sch8 * 8 };
  __bf16* vdst[2] = { vbuf[0] + shd * 32 + sch4 * 8,  vbuf[1] + shd * 32 + sch4 * 8 };

  // Q A-fragments: 2 sub-tiles x 2 K-chunks, resident for the whole loop.
  bfrag qA[2][2];
#pragma unroll
  for (int tq = 0; tq < 2; ++tq) {
    const __bf16* qrow = Q + (bhT + q0 + tq * 16 + lanec) * HD_;
    qA[tq][0] = frag_from_bf16(qrow + half * 8,      qrow + 16 + half * 8);
    qA[tq][1] = frag_from_bf16(qrow + 32 + half * 8, qrow + 48 + half * 8);
  }

  float mrun[2][8], lrun[2][8];
#pragma unroll
  for (int tq = 0; tq < 2; ++tq)
#pragma unroll
    for (int g = 0; g < 8; ++g) { mrun[tq][g] = -1e30f; lrun[tq][g] = 0.0f; }
  v8f O[2][4] = {};

  // Prologue: stage tile 0.
  stage16(ksrc, kdst[0]);
  stage16(vsrc, vdst[0]);
  stage_sync();

  for (int kt = 0; kt <= ktmax; ++kt) {
    const int cur = kt & 1;

    // Issue async staging of tile kt+1 into the other buffer.
    if (kt < ktmax) {
      stage16(ksrc + (size_t)(kt + 1) * 32 * HD_, kdst[cur ^ 1]);
      stage16(vsrc + (size_t)(kt + 1) * 32,       vdst[cur ^ 1]);
    }

    if (kt <= qtw) {                   // wave-uniform causal guard
      const __bf16* kb = kbuf[cur];
      const __bf16* vb = vbuf[cur];

      // K fragments: [0,1] = key cols 0-15 (dc 0,1); [2,3] = key cols 16-31.
      bfrag kf[4], vf[4];
#pragma unroll
      for (int dc = 0; dc < 2; ++dc) {
        const __bf16* kr0 = kb + lanec * HD_ + dc * 32;
        const __bf16* kr1 = kb + (16 + lanec) * HD_ + dc * 32;
        kf[dc]     = frag_from_bf16(kr0 + half * 8, kr0 + 16 + half * 8);
        kf[2 + dc] = frag_from_bf16(kr1 + half * 8, kr1 + 16 + half * 8);
      }
#pragma unroll
      for (int nc = 0; nc < 4; ++nc) {
        const __bf16* vr = vb + (nc * 16 + lanec) * 32;
        vf[nc] = frag_from_bf16(vr + half * 8, vr + 16 + half * 8);
      }

      const bool needmask = (kt == qtw);

#pragma unroll
      for (int tq = 0; tq < 2; ++tq) {
        v8f s0 = {}, s1 = {};
        s0 = WMMA_BF16(qA[tq][0], kf[0], s0);
        s0 = WMMA_BF16(qA[tq][1], kf[1], s0);
        s1 = WMMA_BF16(qA[tq][0], kf[2], s1);
        s1 = WMMA_BF16(qA[tq][1], kf[3], s1);

#pragma unroll
        for (int g = 0; g < 8; ++g) {
          const int qrowi = q0 + tq * 16 + rowb + g;
          float a = s0[g] * SCALE_;
          float b = s1[g] * SCALE_;
          if (needmask) {
            if (kt * 32 + lanec > qrowi)      a = -1e30f;
            if (kt * 32 + 16 + lanec > qrowi) b = -1e30f;
          }
          s0[g] = a; s1[g] = b;
        }

        float corr[8];
#pragma unroll
        for (int g = 0; g < 8; ++g) {
          float rm = redmax16(fmaxf(s0[g], s1[g]));
          float mn = fmaxf(mrun[tq][g], rm);
          corr[g]  = __expf(mrun[tq][g] - mn);
          mrun[tq][g] = mn;
        }
#pragma unroll
        for (int g = 0; g < 8; ++g) {
          s0[g] = __expf(s0[g] - mrun[tq][g]);
          s1[g] = __expf(s1[g] - mrun[tq][g]);
          float rs = redsum16(s0[g] + s1[g]);
          lrun[tq][g] = lrun[tq][g] * corr[g] + rs;
        }
#pragma unroll
        for (int nc = 0; nc < 4; ++nc)
#pragma unroll
          for (int g = 0; g < 8; ++g) O[tq][nc][g] *= corr[g];

        // P: C-layout -> LDS -> A-layout (bf16)
#pragma unroll
        for (int g = 0; g < 8; ++g) {
          sm[(rowb + g) * 32 + lanec]      = (__bf16)s0[g];
          sm[(rowb + g) * 32 + 16 + lanec] = (__bf16)s1[g];
        }
        bfrag pA = frag_from_bf16(sm + lanec * 32 + half * 8,
                                  sm + lanec * 32 + 16 + half * 8);

#pragma unroll
        for (int nc = 0; nc < 4; ++nc)
          O[tq][nc] = WMMA_BF16(pA, vf[nc], O[tq][nc]);
      }
    }

    stage_sync();   // async copies done + all waves finished reading buffers
  }

  // Normalize and store [b, t, h*64+hd] as bf16.
  const int bb = bh >> 4, h = bh & 15;
#pragma unroll
  for (int tq = 0; tq < 2; ++tq) {
#pragma unroll
    for (int g = 0; g < 8; ++g) {
      const float inv = 1.0f / lrun[tq][g];
      const size_t base = (size_t)(bb * T_ + q0 + tq * 16 + rowb + g) * D_ + h * HD_;
#pragma unroll
      for (int nc = 0; nc < 4; ++nc)
        Oout[base + nc * 16 + lanec] = (__bf16)(O[tq][nc][g] * inv);
    }
  }
}

// ---------------------------------------------------------------------------
// Output projection: out = A @ Wo^T + bo   (A,Wo bf16 -> out f32)
// ---------------------------------------------------------------------------
__global__ void out_proj_kernel(const __bf16* __restrict__ A,
                                const __bf16* __restrict__ Wo,
                                const float* __restrict__ bo,
                                float* __restrict__ out) {
  const int lane  = threadIdx.x & 31;
  const int lanec = lane & 15;
  const int half  = lane >> 4;
  const int rowb  = half * 8;

  const int wid   = blockIdx.x * 8 + (threadIdx.x >> 5);   // 4096 wave tasks
  const int n128  = wid & 7;
  const int mtile = wid >> 3;
  const int row0  = mtile * 16;

  v8f acc[8] = {};
  const __bf16* arow  = A + (size_t)(row0 + lanec) * D_;
  const __bf16* wbase = Wo + (size_t)(n128 * 128 + lanec) * D_;

  for (int kb = 0; kb < D_ / 32; ++kb) {
    const int kbase = kb * 32;
    bfrag a = frag_from_bf16(arow + kbase + half * 8,
                             arow + kbase + 16 + half * 8);
    bfrag bf[8];
#pragma unroll
    for (int nt = 0; nt < 8; ++nt) {
      const __bf16* wrow = wbase + (size_t)nt * 16 * D_ + kbase;
      bf[nt] = frag_from_bf16(wrow + half * 8, wrow + 16 + half * 8);
    }
#pragma unroll
    for (int nt = 0; nt < 8; ++nt)
      acc[nt] = WMMA_BF16(a, bf[nt], acc[nt]);
  }

#pragma unroll
  for (int nt = 0; nt < 8; ++nt) {
    const int nf = n128 * 128 + nt * 16 + lanec;
    const float bias = bo[nf];
#pragma unroll
    for (int g = 0; g < 8; ++g) {
      out[(size_t)(row0 + rowb + g) * D_ + nf] = acc[nt][g] + bias;
    }
  }
}

// ---------------------------------------------------------------------------
extern "C" void kernel_launch(void* const* d_in, const int* in_sizes, int n_in,
                              void* d_out, int out_size, void* d_ws, size_t ws_size,
                              hipStream_t stream) {
  (void)in_sizes; (void)n_in; (void)out_size; (void)ws_size;
  const float* querys = (const float*)d_in[0];
  const float* keys   = (const float*)d_in[1];
  const float* values = (const float*)d_in[2];
  const float* Wq     = (const float*)d_in[3];
  const float* Wk     = (const float*)d_in[4];
  const float* Wv     = (const float*)d_in[5];
  const float* Wo     = (const float*)d_in[6];
  const float* bo     = (const float*)d_in[7];
  float* out = (float*)d_out;

  // Workspace: Qb,Kb,Vt,Ao 16 MiB each ; Xs 16 MiB ; Wsb 2 MiB  => 82 MiB.
  const size_t QKV_BYTES = (size_t)B_ * H_ * T_ * HD_ * sizeof(__bf16); // 16 MiB
  char* ws = (char*)d_ws;
  __bf16* Qb  = (__bf16*)(ws);
  __bf16* Kb  = (__bf16*)(ws + QKV_BYTES);
  __bf16* Vt  = (__bf16*)(ws + 2 * QKV_BYTES);
  __bf16* Ao  = (__bf16*)(ws + 3 * QKV_BYTES);
  __bf16* Xs  = (__bf16*)(ws + 4 * QKV_BYTES);
  __bf16* Wsb = (__bf16*)(ws + 5 * QKV_BYTES);

  const int nX8 = (B_ * T_ * D_) / 8;
  const int nW8 = (D_ * D_) / 8;
  dim3 cblkX((nX8 + 255) / 256), cblkW((nW8 + 255) / 256), cthr(256);
  dim3 pgrid(512), agrid(512), block(256);

  f32_to_bf16_kernel<<<cblkX, cthr, 0, stream>>>(querys, Xs, nX8);
  f32_to_bf16_kernel<<<cblkW, cthr, 0, stream>>>(Wq, Wsb, nW8);
  qkv_proj_kernel<<<pgrid, block, 0, stream>>>(Xs, Wsb, Qb, 0);

  f32_to_bf16_kernel<<<cblkX, cthr, 0, stream>>>(keys, Xs, nX8);
  f32_to_bf16_kernel<<<cblkW, cthr, 0, stream>>>(Wk, Wsb, nW8);
  qkv_proj_kernel<<<pgrid, block, 0, stream>>>(Xs, Wsb, Kb, 0);

  f32_to_bf16_kernel<<<cblkX, cthr, 0, stream>>>(values, Xs, nX8);
  f32_to_bf16_kernel<<<cblkW, cthr, 0, stream>>>(Wv, Wsb, nW8);
  qkv_proj_kernel<<<pgrid, block, 0, stream>>>(Xs, Wsb, Vt, 1);

  flash_attn_kernel<<<agrid, block, 0, stream>>>(Qb, Kb, Vt, Ao);

  f32_to_bf16_kernel<<<cblkW, cthr, 0, stream>>>(Wo, Wsb, nW8);
  out_proj_kernel<<<pgrid, block, 0, stream>>>(Ao, Wsb, bo, out);
}
